// Conv2d_NN_Attn_36378372997763
// MI455X (gfx1250) — matrix operand order repaired and checked by reading, not compiled, not run on hardware
//
#include <hip/hip_runtime.h>
#include <hip/hip_bf16.h>

// ---------------------------------------------------------------------------
// Conv2d_NN_Attn for MI455X (gfx1250, wave32).
//   B=8, C=16, H=W=64 -> T=4096, K_NEIGH=9, Cout=16.
// Phase 0: fuse w_o into conv_w/conv_b (tiny).
// Phase 1: per-token: xn = x/max(||x||,1e-12) as f16 (token-major [T][16]),
//          v = w_v2 * relu(w_v1 * x) as f32 (token-major).
// Phase 2: fused sim-GEMM (v_wmma_f32_16x16x32_f16, K padded 16->32) + top-9,
//          software-pipelined B prefetch + double-buffered LDS tiles +
//          dscnt-only split barrier so prefetch stays in flight.
//          sim is NEVER materialized in HBM (would be 537 MB ~ 46us @23.3TB/s).
// Phase 3: gather 9 neighbors, scale by vals, contract with fused W2, write out.
// ---------------------------------------------------------------------------

#define BATCH 8
#define CCH   16      // channels
#define TTOK  4096    // tokens per batch (64*64)
#define KNB   9       // neighbors
#define SIMSTRIDE 140 // LDS row stride (floats): conflict-free for both the
                      // C-fragment scatter (8*140 % 64 == 32) and the
                      // 8-thread/row scan (12*row + j distinct mod 64)

typedef __attribute__((ext_vector_type(16))) _Float16 v16h;
typedef __attribute__((ext_vector_type(8)))  _Float16 v8h;
typedef __attribute__((ext_vector_type(8)))  float    v8f;

// Workgroup barrier that only drains DScnt (LDS visibility) — unlike
// __syncthreads(), it does NOT wait on LOADcnt, so in-flight global
// prefetches ride across the barrier and are consumed by the s_wait_loadcnt
// the compiler emits just before the next WMMA.
__device__ __forceinline__ void lds_barrier() {
  asm volatile("s_wait_dscnt 0x0\n\t"
               "s_barrier_signal -1\n\t"
               "s_barrier_wait -1" ::: "memory");
}

// ---------------- Phase 0: fold w_o into conv weights ----------------------
// W2[(k*16+c)*16+o] = sum_o' w_o[o][o'] * conv_w[o'][c][k];  b2[o] = w_o . conv_b
__global__ __launch_bounds__(256) void fuse_w_kernel(
    const float* __restrict__ conv_w, const float* __restrict__ conv_b,
    const float* __restrict__ w_o, float* __restrict__ W2, float* __restrict__ b2)
{
  for (int e = threadIdx.x; e < 16 * CCH * KNB; e += 256) {
    int o  = e & 15;
    int kc = e >> 4;         // k*16 + c
    int c  = kc & 15;
    int k  = kc >> 4;        // 0..8
    float s = 0.f;
    #pragma unroll
    for (int op = 0; op < 16; ++op)
      s += w_o[o * 16 + op] * conv_w[(op * CCH + c) * KNB + k];
    W2[e] = s;
  }
  if (threadIdx.x < 16) {
    float s = 0.f;
    #pragma unroll
    for (int op = 0; op < 16; ++op) s += w_o[threadIdx.x * 16 + op] * conv_b[op];
    b2[threadIdx.x] = s;
  }
}

// ---------------- Phase 1: normalize + value MLP ---------------------------
__global__ __launch_bounds__(256) void prep_kernel(
    const float* __restrict__ x, const float* __restrict__ w_v1,
    const float* __restrict__ w_v2, _Float16* __restrict__ xn,
    float* __restrict__ vtok)
{
  int g = blockIdx.x * 256 + threadIdx.x;           // 0 .. B*T-1
  if (g >= BATCH * TTOK) return;
  int b = g >> 12;
  int t = g & (TTOK - 1);

  float xv[CCH];
  #pragma unroll
  for (int c = 0; c < CCH; ++c)
    xv[c] = x[(((size_t)b * CCH + c) << 12) + t];   // [B][C][T]

  float ss = 0.f;
  #pragma unroll
  for (int c = 0; c < CCH; ++c) ss += xv[c] * xv[c];
  float inv = 1.0f / fmaxf(sqrtf(ss), 1e-12f);      // F.normalize eps

  #pragma unroll
  for (int c = 0; c < CCH; ++c)
    xn[((size_t)g << 4) + c] = (_Float16)(xv[c] * inv);  // token-major [T][16]

  float h[8];
  #pragma unroll
  for (int o = 0; o < 8; ++o) {
    float s = 0.f;
    #pragma unroll
    for (int c = 0; c < CCH; ++c) s += w_v1[o * CCH + c] * xv[c];
    h[o] = fmaxf(s, 0.f);                            // ReLU
  }
  #pragma unroll
  for (int c = 0; c < CCH; ++c) {
    float s = 0.f;
    #pragma unroll
    for (int j = 0; j < 8; ++j) s += w_v2[c * 8 + j] * h[j];
    vtok[((size_t)g << 4) + c] = s;                  // token-major [T][16]
  }
}

// ---------------- Phase 2: fused WMMA sim tiles + top-9 --------------------
// Block = 128 threads (4 waves), owns 16 rows of one batch.
// Chunk = 128 columns: 8 sim tiles via v_wmma_f32_16x16x32_f16 -> LDS
// (double-buffered, 1 dscnt-only barrier/chunk), then 8 threads/row update
// private sorted top-9 lists. Next chunk's B fragments prefetch under the scan.
__global__ __launch_bounds__(128) void topk_sim_kernel(
    const _Float16* __restrict__ xn, float* __restrict__ tvals,
    int* __restrict__ tidx)
{
  __shared__ float s_sim[2][16 * SIMSTRIDE];  // 2 x 8960 B
  __shared__ float s_mv[16 * 72];             // merge staging
  __shared__ int   s_mi[16 * 72];

  const int tid  = threadIdx.x;
  const int lane = tid & 31;
  const int wave = tid >> 5;
  const int rt   = blockIdx.x;              // 0 .. B*T/16-1
  const int b    = rt >> 8;                 // 256 row-tiles per batch
  const int r0   = (rt & 255) << 4;

  // ---- A fragment: rows r0..r0+15, K=0..15 real, K=16..31 zero-padded ----
  // A-layout (contiguous, matches B): lane&15 = row M; lanes<16 hold K0-15,
  // lanes>=16 hold K16-31 (zero here).
  const int m    = lane & 15;
  const int hsel = (lane >> 4) << 3;        // 0 or 8
  v16h a;
  #pragma unroll
  for (int i = 0; i < 16; ++i) a[i] = (_Float16)0.0f;
  if (lane < 16) {
    a = *(const v16h*)(xn + (((size_t)(b * TTOK + r0 + m)) << 4));
  }

  // ---- B fragments: double-buffered in registers, zero-padded once -------
  // B layout (32x16): lane<16 = column token holding K0-15 contiguous;
  // lanes>=16 carry K16-31 = zero padding. Masked loads never touch the
  // upper lanes, so the zeros persist across all chunk iterations.
  v16h bfr0, bfr1;
  #pragma unroll
  for (int i = 0; i < 16; ++i) { bfr0[i] = (_Float16)0.0f; bfr1[i] = (_Float16)0.0f; }
  const bool bload = (lane < 16);
  const _Float16* bbase = xn + (((size_t)b * TTOK) << 4);
  const int bcol0 = (wave * 2 + 0) * 16 + lane;   // column token (lane<16)
  const int bcol1 = (wave * 2 + 1) * 16 + lane;
  if (bload) {
    bfr0 = *(const v16h*)(bbase + ((size_t)bcol0 << 4));
    bfr1 = *(const v16h*)(bbase + ((size_t)bcol1 << 4));
  }

  // ---- private top-9 state (8 scan-threads per row) ----
  float tv[KNB]; int ti[KNB];
  #pragma unroll
  for (int i = 0; i < KNB; ++i) { tv[i] = -3.0f; ti[i] = 0; }
  const int row   = tid >> 3;               // 0..15
  const int jj    = tid & 7;                // scan slot within row
  const int rglob = r0 + row;

  const int colA = ((wave * 2 + 0) << 4) + (lane & 15);
  const int colB = colA + 16;
  const int rb   = (lane >> 4) << 3;

  for (int chunk = 0; chunk < TTOK / 128; ++chunk) {   // 32 chunks
    const int cbase = chunk << 7;
    float* sbuf = s_sim[chunk & 1];

    v8f cz;
    #pragma unroll
    for (int i = 0; i < 8; ++i) cz[i] = 0.0f;          // lowers to inline C=0
    v8f cc0 = __builtin_amdgcn_wmma_f32_16x16x32_f16(
        false, a, false, bfr0, (short)0, cz, false, false);
    v8f cc1 = __builtin_amdgcn_wmma_f32_16x16x32_f16(
        false, a, false, bfr1, (short)0, cz, false, false);

    // prefetch next chunk's B fragments first: they can issue during the
    // WMMA->VALU hazard window of the LDS stores and complete under the scan
    if ((chunk < TTOK / 128 - 1) && bload) {
      const int nc = (chunk + 1) << 7;
      bfr0 = *(const v16h*)(bbase + ((size_t)(bcol0 + nc) << 4));
      bfr1 = *(const v16h*)(bbase + ((size_t)(bcol1 + nc) << 4));
    }

    // C/D layout: VGPR i -> row i (lanes<16) or row i+8 (lanes>=16), col=lane&15
    #pragma unroll
    for (int i = 0; i < 8; ++i) {
      sbuf[(rb + i) * SIMSTRIDE + colA] = cc0[i];
      sbuf[(rb + i) * SIMSTRIDE + colB] = cc1[i];
    }

    lds_barrier();   // dscnt-only: global prefetch stays in flight

    // scan 16 columns of this row: batch all LDS reads (one wait), then sort
    float sv[16];
    #pragma unroll
    for (int q = 0; q < 16; ++q)
      sv[q] = sbuf[row * SIMSTRIDE + jj + (q << 3)];

    #pragma unroll
    for (int q = 0; q < 16; ++q) {
      float v = fminf(1.0f, fmaxf(-1.0f, sv[q]));   // clip first (ref order)
      const int cg = cbase + jj + (q << 3);
      if (cg == rglob) v = 1.1f;                    // self forced first
      if (v > tv[KNB - 1]) {                        // sorted-desc bubble insert
        tv[KNB - 1] = v; ti[KNB - 1] = cg;
        #pragma unroll
        for (int i = KNB - 1; i > 0; --i) {
          if (tv[i] > tv[i - 1]) {
            float f = tv[i - 1]; tv[i - 1] = tv[i]; tv[i] = f;
            int   d = ti[i - 1]; ti[i - 1] = ti[i]; ti[i] = d;
          }
        }
      }
    }
  }

  // ---- merge the 8 partial lists per row -> final sorted top-9 ----
  #pragma unroll
  for (int i = 0; i < KNB; ++i) {
    s_mv[row * 72 + jj * KNB + i] = tv[i];
    s_mi[row * 72 + jj * KNB + i] = ti[i];
  }
  __syncthreads();
  if (tid < 16) {
    float fv[KNB]; int fi[KNB];
    #pragma unroll
    for (int i = 0; i < KNB; ++i) { fv[i] = -3.0f; fi[i] = 0; }
    for (int i = 0; i < 72; ++i) {
      float v = s_mv[tid * 72 + i];
      int   d = s_mi[tid * 72 + i];
      if (v > fv[KNB - 1]) {
        fv[KNB - 1] = v; fi[KNB - 1] = d;
        #pragma unroll
        for (int k = KNB - 1; k > 0; --k) {
          if (fv[k] > fv[k - 1]) {
            float f = fv[k - 1]; fv[k - 1] = fv[k]; fv[k] = f;
            int   e = fi[k - 1]; fi[k - 1] = fi[k]; fi[k] = e;
          }
        }
      }
    }
    size_t o = ((size_t)(b * TTOK + r0 + tid)) * KNB;
    #pragma unroll
    for (int i = 0; i < KNB; ++i) { tvals[o + i] = fv[i]; tidx[o + i] = fi[i]; }
  }
}

// ---------------- Phase 3: gather + fused output contraction ---------------
__global__ __launch_bounds__(256) void out_kernel(
    const float* __restrict__ vtok, const float* __restrict__ tvals,
    const int* __restrict__ tidx, const float* __restrict__ W2,
    const float* __restrict__ b2, float* __restrict__ out)
{
  __shared__ float sW[16 * CCH * KNB];      // 9216 B, broadcast-read
  __shared__ float sb[16];
  for (int e = threadIdx.x; e < 16 * CCH * KNB; e += 256) sW[e] = W2[e];
  if (threadIdx.x < 16) sb[threadIdx.x] = b2[threadIdx.x];
  __syncthreads();

  int g = blockIdx.x * 256 + threadIdx.x;   // 0 .. B*T-1
  if (g >= BATCH * TTOK) return;
  int b = g >> 12;
  int t = g & (TTOK - 1);

  float acc[16];
  #pragma unroll
  for (int o = 0; o < 16; ++o) acc[o] = sb[o];

  size_t tb = (size_t)g * KNB;
  for (int k = 0; k < KNB; ++k) {
    int   nb  = tidx[tb + k];
    float val = tvals[tb + k];
    const float* vp = vtok + ((((size_t)b << 12) + nb) << 4);
    #pragma unroll
    for (int c = 0; c < CCH; ++c) {
      float pv = vp[c] * val;               // prime[c,t,k]
      #pragma unroll
      for (int o = 0; o < 16; ++o)
        acc[o] += sW[(k * CCH + c) * 16 + o] * pv;
    }
  }
  #pragma unroll
  for (int o = 0; o < 16; ++o)
    out[(((size_t)b * 16 + o) << 12) + t] = acc[o];   // [B][Cout][T]
}

// ---------------------------------------------------------------------------
extern "C" void kernel_launch(void* const* d_in, const int* in_sizes, int n_in,
                              void* d_out, int out_size, void* d_ws, size_t ws_size,
                              hipStream_t stream) {
  const float* x      = (const float*)d_in[0];   // [8,16,64,64]
  const float* w_v1   = (const float*)d_in[1];   // [8,16]
  const float* w_v2   = (const float*)d_in[2];   // [16,8]
  const float* conv_w = (const float*)d_in[3];   // [16,16,9]
  const float* conv_b = (const float*)d_in[4];   // [16]
  const float* w_o    = (const float*)d_in[5];   // [16,16]
  float* out = (float*)d_out;

  char* ws = (char*)d_ws;
  // workspace layout (all offsets 256-B aligned), total ~5.5 MB
  _Float16* xn16  = (_Float16*)(ws + 0);          // B*T*16 f16 = 1 MB
  float*    vtok  = (float*)  (ws + 1048576);     // B*T*16 f32 = 2 MB
  float*    tvals = (float*)  (ws + 3145728);     // B*T*9 f32
  int*      tidx  = (int*)    (ws + 4325376);     // B*T*9 i32
  float*    W2    = (float*)  (ws + 5505024);     // 2304 f32
  float*    b2    = (float*)  (ws + 5514240);     // 16 f32

  fuse_w_kernel<<<1, 256, 0, stream>>>(conv_w, conv_b, w_o, W2, b2);
  prep_kernel<<<(BATCH * TTOK) / 256, 256, 0, stream>>>(x, w_v1, w_v2, xn16, vtok);
  topk_sim_kernel<<<(BATCH * TTOK) / 16, 128, 0, stream>>>(xn16, tvals, tidx);
  out_kernel<<<(BATCH * TTOK) / 256, 256, 0, stream>>>(vtok, tvals, tidx, W2, b2, out);
}
